// WindowedAttention_463856468479
// MI455X (gfx1250) — matrix operand order repaired and checked by reading, not compiled, not run on hardware
//
#include <hip/hip_runtime.h>

// ---------------------------------------------------------------------------
// MI455X (gfx1250) windowed attention, bf16 WMMA pipeline.
// - All matmuls: v_wmma_f32_16x16x32_bf16 (f32 accumulate)
// - Attention: flash-style online softmax, V tile staged to LDS via
//   global_load_async_to_lds_b128 (ASYNCcnt path), K^T staged via LDS transpose
// - GEMMs: 32x64 wave tiles (B fragments reused across 2 WMMAs)
// Workspace (bf16): x_pad 51.4MB | qkv 154.1MB | attn_out 51.4MB | weights 0.5MB
// ---------------------------------------------------------------------------

typedef __attribute__((ext_vector_type(16))) __bf16 v16bf;
typedef __attribute__((ext_vector_type(8)))  __bf16 v8bf;
typedef __attribute__((ext_vector_type(8)))  float  v8f;

#define NB    2
#define NSEQ  50000
#define CDIM  256
#define NHEAD 8
#define HD    32
#define WIN   256
#define NPAD  50176              // per-batch padded rows
#define MTOT  100352             // NB * NPAD

__device__ __forceinline__ v8f zero8() {
  v8f z;
#pragma unroll
  for (int i = 0; i < 8; ++i) z[i] = 0.0f;
  return z;
}

__device__ __forceinline__ v16bf frag_cat(v8bf lo, v8bf hi) {
  return __builtin_shufflevector(lo, hi, 0,1,2,3,4,5,6,7,8,9,10,11,12,13,14,15);
}

// A fragment (16x32 bf16): lane = M row (l&15), k halves split by l>>4.
// elements 0..7 <- k = 8*(l>>4)+0..7 ; elements 8..15 <- k = 16+8*(l>>4)+0..7
__device__ __forceinline__ v16bf load_a(const __bf16* base, int ld, int lane) {
  const __bf16* p = base + (size_t)(lane & 15) * ld + ((lane >> 4) << 3);
  return frag_cat(*(const v8bf*)p, *(const v8bf*)(p + 16));
}

// B fragment (32x16 bf16): lane = N col (l&15); elements 0..7 <- k = 8*(l>>4)+0..7 ; 8..15 <- k = 16+8*(l>>4)+0..7
__device__ __forceinline__ v16bf load_b_g(const __bf16* base, int ld, int lane) {
  const __bf16* p = base + (lane & 15) + (size_t)((lane >> 4) << 3) * ld;
  v16bf r;
#pragma unroll
  for (int s = 0; s < 16; ++s) r[s] = p[(size_t)(s + ((s >> 3) << 3)) * ld];
  return r;
}

__device__ __forceinline__ v16bf load_b_s(const __bf16* base, int ld, int lane) {
  const __bf16* p = base + (lane & 15) + ((lane >> 4) << 3) * ld;
  v16bf r;
#pragma unroll
  for (int s = 0; s < 16; ++s) r[s] = p[(s + ((s >> 3) << 3)) * ld];
  return r;
}

__device__ __forceinline__ v8f wmma_bf16(v16bf a, v16bf b, v8f c) {
  return __builtin_amdgcn_wmma_f32_16x16x32_bf16(false, a, false, b,
                                                 (short)0, c, false, false);
}

// --------------------------- prep kernels ----------------------------------

__global__ void cvt_x_kernel(const float* __restrict__ x, __bf16* __restrict__ xb) {
  int m = blockIdx.x;                 // 0..MTOT-1
  int c = threadIdx.x;                // 0..255
  int b = m / NPAD;
  int r = m - b * NPAD;
  float v = 0.0f;
  if (r < NSEQ) v = x[((size_t)b * NSEQ + r) * CDIM + c];
  xb[(size_t)m * CDIM + c] = (__bf16)v;
}

__global__ void cvt_w_kernel(const float* __restrict__ w, __bf16* __restrict__ wb, int n) {
  int i = blockIdx.x * blockDim.x + threadIdx.x;
  if (i < n) wb[i] = (__bf16)w[i];
}

// --------------------------- QKV GEMM --------------------------------------
// grid (392, 12), block 256 (8 waves). Wave tile: 32(M) x 64(N), K=256.

__global__ __launch_bounds__(256) void qkv_gemm_kernel(
    const __bf16* __restrict__ X, const __bf16* __restrict__ W,
    const float* __restrict__ bias, __bf16* __restrict__ out) {
  int wv = threadIdx.x >> 5;
  int lane = threadIdx.x & 31;
  int m0 = blockIdx.x * 256 + wv * 32;
  int n0 = blockIdx.y * 64;
  v8f acc[2][4];
#pragma unroll
  for (int mt = 0; mt < 2; ++mt)
#pragma unroll
    for (int j = 0; j < 4; ++j) acc[mt][j] = zero8();

#pragma unroll
  for (int k0 = 0; k0 < 256; k0 += 32) {
    v16bf a0 = load_a(X + (size_t)m0 * 256 + k0, 256, lane);
    v16bf a1 = load_a(X + (size_t)(m0 + 16) * 256 + k0, 256, lane);
#pragma unroll
    for (int j = 0; j < 4; ++j) {
      v16bf b = load_b_g(W + (size_t)k0 * 768 + n0 + j * 16, 768, lane);
      acc[0][j] = wmma_bf16(a0, b, acc[0][j]);
      acc[1][j] = wmma_bf16(a1, b, acc[1][j]);
    }
  }
  int half = lane >> 4;
#pragma unroll
  for (int mt = 0; mt < 2; ++mt)
#pragma unroll
    for (int j = 0; j < 4; ++j) {
      int col = n0 + j * 16 + (lane & 15);
      float bb = bias[col];
#pragma unroll
      for (int v = 0; v < 8; ++v) {
        int row = m0 + mt * 16 + v + (half << 3);
        out[(size_t)row * 768 + col] = (__bf16)(acc[mt][j][v] + bb);
      }
    }
}

// --------------------------- Attention -------------------------------------
// One block per (window, head). 8 waves; wave owns 32 rows. Flash softmax.

__global__ __launch_bounds__(256) void attn_kernel(
    const __bf16* __restrict__ qkv, __bf16* __restrict__ aout) {
  __shared__ __bf16 kt[32 * 264];     // k transposed [d][j], padded stride
  __shared__ __bf16 pw[8][32][40];    // per-wave P bounce buffer, padded
  __shared__ __bf16 vt[256 * 40];     // v tile [j][d], padded stride (80B rows)

  int h   = blockIdx.x & 7;
  int win = blockIdx.x >> 3;
  int rb  = win * 256;                // row base in MTOT space
  int t   = threadIdx.x;

  { // stage V tile (256x32) into LDS via async copies (ASYNCcnt path)
    const __bf16* vbase = qkv + (size_t)rb * 768 + 512 + h * 32;
#pragma unroll
    for (int i = 0; i < 4; ++i) {
      int c = t + i * 256;            // 16B chunk id, 0..1023
      int j = c >> 2;
      int part = c & 3;
      unsigned ldsa = (unsigned)(uintptr_t)&vt[j * 40 + part * 8];
      unsigned long long ga =
          (unsigned long long)(uintptr_t)(vbase + (size_t)j * 768 + part * 8);
      asm volatile("global_load_async_to_lds_b128 %0, %1, off"
                   :: "v"(ldsa), "v"(ga) : "memory");
    }
  }
  { // stage k^T into LDS: kt[d][j] = k[j][d]
    const __bf16* kr = qkv + (size_t)(rb + t) * 768 + 256 + h * 32;
    v16bf klo = *(const v16bf*)kr;
    v16bf khi = *(const v16bf*)(kr + 16);
#pragma unroll
    for (int d = 0; d < 16; ++d) {
      kt[d * 264 + t]        = klo[d];
      kt[(d + 16) * 264 + t] = khi[d];
    }
  }
  asm volatile("s_wait_asynccnt 0" ::: "memory");
  __syncthreads();

  int wv = t >> 5, lane = t & 31, half = lane >> 4;
  const float scale = 0.1767766952966369f;   // HD^-0.5

  v8f o[2][2];
  float rmax[2][8], rsum[2][8];
#pragma unroll
  for (int mt = 0; mt < 2; ++mt) {
    o[mt][0] = zero8(); o[mt][1] = zero8();
#pragma unroll
    for (int v = 0; v < 8; ++v) { rmax[mt][v] = -1e30f; rsum[mt][v] = 0.0f; }
  }

  v16bf aq[2];
#pragma unroll
  for (int mt = 0; mt < 2; ++mt)
    aq[mt] = load_a(qkv + (size_t)(rb + wv * 32 + mt * 16) * 768 + h * 32, 768, lane);

  for (int j0 = 0; j0 < 256; j0 += 32) {
    // S chunk = q (32x32) @ k^T (32x32-cols)
    v16bf bk0 = load_b_s(kt + j0, 264, lane);
    v16bf bk1 = load_b_s(kt + j0 + 16, 264, lane);
    v8f s[2][2];
#pragma unroll
    for (int mt = 0; mt < 2; ++mt) {
      s[mt][0] = wmma_bf16(aq[mt], bk0, zero8());
      s[mt][1] = wmma_bf16(aq[mt], bk1, zero8());
    }
    // online softmax update; write P (bf16) to per-wave LDS bounce buffer
#pragma unroll
    for (int mt = 0; mt < 2; ++mt) {
#pragma unroll
      for (int v = 0; v < 8; ++v) {
        float s0 = s[mt][0][v] * scale;
        float s1 = s[mt][1][v] * scale;
        float cm = fmaxf(s0, s1);
#pragma unroll
        for (int msk = 1; msk < 16; msk <<= 1)
          cm = fmaxf(cm, __shfl_xor(cm, msk, 32));
        float nm = fmaxf(rmax[mt][v], cm);
        float cr = __expf(rmax[mt][v] - nm);
        float p0 = __expf(s0 - nm);
        float p1 = __expf(s1 - nm);
        float cs = p0 + p1;
#pragma unroll
        for (int msk = 1; msk < 16; msk <<= 1)
          cs += __shfl_xor(cs, msk, 32);
        rsum[mt][v] = rsum[mt][v] * cr + cs;
        rmax[mt][v] = nm;
        o[mt][0][v] *= cr;
        o[mt][1][v] *= cr;
        int ml = mt * 16 + v + (half << 3);
        pw[wv][ml][lane & 15]        = (__bf16)p0;
        pw[wv][ml][16 + (lane & 15)] = (__bf16)p1;
      }
    }
    // O += P (32x32) @ v (32x32), V fragments from LDS
    v16bf ap0 = load_a(&pw[wv][0][0], 40, lane);
    v16bf ap1 = load_a(&pw[wv][16][0], 40, lane);
#pragma unroll
    for (int dt = 0; dt < 2; ++dt) {
      v16bf bv = load_b_s(vt + j0 * 40 + dt * 16, 40, lane);
      o[0][dt] = wmma_bf16(ap0, bv, o[0][dt]);
      o[1][dt] = wmma_bf16(ap1, bv, o[1][dt]);
    }
  }

  // normalize and store
#pragma unroll
  for (int mt = 0; mt < 2; ++mt) {
#pragma unroll
    for (int v = 0; v < 8; ++v) {
      float inv = 1.0f / rsum[mt][v];
      int row = rb + wv * 32 + mt * 16 + v + (half << 3);
#pragma unroll
      for (int dt = 0; dt < 2; ++dt)
        aout[(size_t)row * 256 + h * 32 + dt * 16 + (lane & 15)] =
            (__bf16)(o[mt][dt][v] * inv);
    }
  }
}

// --------------------------- Proj GEMM -------------------------------------
// grid (392, 4). attn_out (100352x256) @ proj_w (256x256) + b -> f32 out

__global__ __launch_bounds__(256) void proj_gemm_kernel(
    const __bf16* __restrict__ X, const __bf16* __restrict__ W,
    const float* __restrict__ bias, float* __restrict__ out) {
  int wv = threadIdx.x >> 5;
  int lane = threadIdx.x & 31;
  int m0 = blockIdx.x * 256 + wv * 32;
  int n0 = blockIdx.y * 64;
  v8f acc[2][4];
#pragma unroll
  for (int mt = 0; mt < 2; ++mt)
#pragma unroll
    for (int j = 0; j < 4; ++j) acc[mt][j] = zero8();

#pragma unroll
  for (int k0 = 0; k0 < 256; k0 += 32) {
    v16bf a0 = load_a(X + (size_t)m0 * 256 + k0, 256, lane);
    v16bf a1 = load_a(X + (size_t)(m0 + 16) * 256 + k0, 256, lane);
#pragma unroll
    for (int j = 0; j < 4; ++j) {
      v16bf b = load_b_g(W + (size_t)k0 * 256 + n0 + j * 16, 256, lane);
      acc[0][j] = wmma_bf16(a0, b, acc[0][j]);
      acc[1][j] = wmma_bf16(a1, b, acc[1][j]);
    }
  }
  int half = lane >> 4;
#pragma unroll
  for (int mt = 0; mt < 2; ++mt)
#pragma unroll
    for (int j = 0; j < 4; ++j) {
      int col = n0 + j * 16 + (lane & 15);
      float bb = bias[col];
#pragma unroll
      for (int v = 0; v < 8; ++v) {
        int row = m0 + mt * 16 + v + (half << 3);
        int b = row / NPAD;
        int r = row - b * NPAD;
        if (r < NSEQ)
          out[((size_t)b * NSEQ + r) * CDIM + col] = acc[mt][j][v] + bb;
      }
    }
}

// --------------------------- launch ----------------------------------------

extern "C" void kernel_launch(void* const* d_in, const int* in_sizes, int n_in,
                              void* d_out, int out_size, void* d_ws, size_t ws_size,
                              hipStream_t stream) {
  (void)in_sizes; (void)n_in; (void)out_size; (void)ws_size;
  const float* x      = (const float*)d_in[0];
  const float* qkv_w  = (const float*)d_in[1];
  const float* qkv_b  = (const float*)d_in[2];
  const float* proj_w = (const float*)d_in[3];
  const float* proj_b = (const float*)d_in[4];
  float* out = (float*)d_out;

  // workspace layout (bf16), ~257.4 MB total, 32B-aligned segments
  __bf16* x_bf    = (__bf16*)d_ws;                  // MTOT*256
  __bf16* qkv_bf  = x_bf   + (size_t)MTOT * 256;    // MTOT*768
  __bf16* attn_bf = qkv_bf + (size_t)MTOT * 768;    // MTOT*256
  __bf16* wq_bf   = attn_bf + (size_t)MTOT * 256;   // 256*768
  __bf16* wp_bf   = wq_bf  + 256 * 768;             // 256*256

  cvt_x_kernel<<<dim3(MTOT), 256, 0, stream>>>(x, x_bf);
  cvt_w_kernel<<<dim3((256 * 768 + 255) / 256), 256, 0, stream>>>(qkv_w, wq_bf, 256 * 768);
  cvt_w_kernel<<<dim3((256 * 256 + 255) / 256), 256, 0, stream>>>(proj_w, wp_bf, 256 * 256);

  qkv_gemm_kernel<<<dim3(MTOT / 256, 768 / 64), 256, 0, stream>>>(x_bf, wq_bf, qkv_b, qkv_bf);
  attn_kernel<<<dim3((MTOT / WIN) * NHEAD), 256, 0, stream>>>(qkv_bf, attn_bf);
  proj_gemm_kernel<<<dim3(MTOT / 256, 256 / 64), 256, 0, stream>>>(attn_bf, wp_bf, proj_b, out);
}
